// Attention_81252191306703
// MI455X (gfx1250) — matrix operand order repaired and checked by reading, not compiled, not run on hardware
//
#include <hip/hip_runtime.h>
#include <hip/hip_bf16.h>

typedef __attribute__((ext_vector_type(16))) __bf16 v16bf;
typedef __attribute__((ext_vector_type(8)))  float  v8f;

// LDS layout (ushort elements), all sub-buffers 16B aligned
#define SW_W 264   // W_t stride (256 K cols + 8 pad)
#define SW_K 264   // k_sh / kz_sh / qz_sh / q_sh stride
#define SW_V 136   // v_t / p_sh stride (128 cols + 8 pad)
#define OFF_WT   0                    // 256*264 = 67584
#define OFF_KV   67584                // max(128*264, 256*136) = 34816
#define OFF_KZ   102400               // 128*264 = 33792
#define OFF_QZ   136192               // 16*264 = 4224
#define OFF_Q    140416               // 16*264 = 4224
#define OFF_P    144640               // 16*136 = 2176
#define OFF_S    146816               // 4*128 f32 = 512 f32 = 1024 ushort
#define OFF_MSK  147840               // 128 bytes = 64 ushort
#define LDS_USHORTS 147904            // 295808 bytes total (< 320KB WGP limit)

__device__ __forceinline__ unsigned short f2bf(float f) {
  union { float f; unsigned int u; } x; x.f = f;
  unsigned int u = x.u;
  unsigned int r = (u + 0x7FFFu + ((u >> 16) & 1u)) >> 16;   // RNE
  return (unsigned short)r;
}

union V16 { uint4 u[2]; v16bf v; };

// A-tile 16x32 bf16 from row-major LDS [rows][stride]:
// lane<16: M=lane, K runs {0..7, 16..23}; lane>=16: M=lane-16, K runs {8..15, 24..31}
__device__ __forceinline__ v16bf lds_tileA(const unsigned short* buf, int stride,
                                           int k0, int lane) {
  int row = lane & 15;
  int kb  = k0 + ((lane >> 4) << 3);
  const unsigned short* p = buf + row * stride + kb;
  V16 t;
  t.u[0] = *(const uint4*)(p);        // e=0..7
  t.u[1] = *(const uint4*)(p + 16);   // e=8..15
  return t.v;
}

// B-tile 32x16 bf16 from N-major LDS [Nrows][stride] (K contiguous):
// lane<16: N=lane, K runs {0..7, 16..23}; lane>=16: N=lane-16, K runs {8..15, 24..31}
__device__ __forceinline__ v16bf lds_tileB(const unsigned short* buf, int stride,
                                           int nbase, int k0, int lane) {
  int row = nbase + (lane & 15);
  int kb  = k0 + ((lane >> 4) << 3);
  const unsigned short* p = buf + row * stride + kb;
  V16 t;
  t.u[0] = *(const uint4*)(p);        // e=0..7
  t.u[1] = *(const uint4*)(p + 16);   // e=8..15
  return t.v;
}

__device__ __forceinline__ v8f wmma_bf16(v16bf a, v16bf b, v8f c) {
  return __builtin_amdgcn_wmma_f32_16x16x32_bf16(false, a, false, b,
                                                 (short)0, c, false, false);
}

__global__ __launch_bounds__(256)
void attn_grouped_kernel(const float* __restrict__ q,
                         const float* __restrict__ k,
                         const float* __restrict__ v,
                         const unsigned char* __restrict__ m,
                         const float* __restrict__ Wq,
                         const float* __restrict__ bq,
                         const float* __restrict__ Wk,
                         const float* __restrict__ bk,
                         float* __restrict__ out) {
  extern __shared__ unsigned short lds[];
  unsigned short* W_t   = lds + OFF_WT;   // [N=256][K(pad 264)] transposed weights
  unsigned short* k_sh  = lds + OFF_KV;   // [l=128][d(pad 264)] row-major
  unsigned short* v_t   = lds + OFF_KV;   // [d=256][l(pad 136)] transposed (reuse)
  unsigned short* kz_sh = lds + OFF_KZ;   // [l=128][d(pad 264)]
  unsigned short* qz_sh = lds + OFF_QZ;   // [16][d(pad 264)]
  unsigned short* q_sh  = lds + OFF_Q;    // [16][d(pad 264)]
  unsigned short* p_sh  = lds + OFF_P;    // [16][l(pad 136)]
  float*          s_sh  = (float*)(lds + OFF_S);            // [4][128]
  unsigned char*  msk   = (unsigned char*)(lds + OFF_MSK);  // [128]

  const int g    = blockIdx.x;
  const int t    = threadIdx.x;
  const int lane = t & 31;
  const int wave = t >> 5;

  const float* kg = k + (size_t)g * 128 * 256;
  const float* vg = v + (size_t)g * 128 * 256;

  // Prefetch this group's k tile while we chew on Wq (global_prefetch_b8).
#pragma unroll
  for (int j = 0; j < 4; ++j)
    __builtin_prefetch(kg + t * 128 + j * 32, 0, 1);

  // ---- A1: Wq -> W_t (transposed, bf16). Coalesced global reads. ----
#pragma unroll 4
  for (int i = 0; i < 256; ++i)
    W_t[t * SW_W + i] = f2bf(Wq[i * 256 + t]);

  // ---- A2: q rows (4 valid) -> q_sh; zero pad rows; zero p_sh; mask. ----
  {
    const float* qg = q + (size_t)g * 4 * 256;
#pragma unroll
    for (int i = 0; i < 4; ++i)  q_sh[i * SW_K + t] = f2bf(qg[i * 256 + t]);
#pragma unroll
    for (int i = 4; i < 16; ++i) q_sh[i * SW_K + t] = 0;
    for (int i = t; i < 16 * SW_V; i += 256) p_sh[i] = 0;
    if (t < 128) msk[t] = m[g * 128 + t];
  }
  __syncthreads();

  // ---- A3: qz = q @ Wq + bq  (M=16 pad, N=256, K=256), 2 N-tiles/wave ----
  {
    v16bf aT[8];                      // A fragment reused across both N-tiles
#pragma unroll
    for (int kt = 0; kt < 8; ++kt) aT[kt] = lds_tileA(q_sh, SW_K, kt * 32, lane);
#pragma unroll
    for (int nn = 0; nn < 2; ++nn) {
      int nt = wave * 2 + nn;
      v8f acc;
#pragma unroll
      for (int r = 0; r < 8; ++r) acc[r] = 0.0f;
#pragma unroll
      for (int kt = 0; kt < 8; ++kt)
        acc = wmma_bf16(aT[kt], lds_tileB(W_t, SW_W, nt * 16, kt * 32, lane), acc);
      int mo = (lane >> 4) << 3;
      int n  = nt * 16 + (lane & 15);
      float bias = bq[n];
#pragma unroll
      for (int r = 0; r < 8; ++r)
        qz_sh[(mo + r) * SW_K + n] = f2bf(acc[r] + bias);
    }
  }
  __syncthreads();

  // ---- B1: Wk -> W_t (overwrite), k[g] -> k_sh (row-major bf16). ----
#pragma unroll 4
  for (int i = 0; i < 256; ++i)
    W_t[t * SW_W + i] = f2bf(Wk[i * 256 + t]);
#pragma unroll 4
  for (int i = 0; i < 128; ++i)
    k_sh[i * SW_K + t] = f2bf(kg[i * 256 + t]);
  __syncthreads();

  // Prefetch v while the big projection matmul runs.
#pragma unroll
  for (int j = 0; j < 4; ++j)
    __builtin_prefetch(vg + t * 128 + j * 32, 0, 1);

  // ---- B2: kz = k @ Wk + bk  (M=128, N=256, K=256); wave w owns M-tile w ----
  // A fragment (this wave's 16 k-rows, full K=256) hoisted into 64 VGPRs and
  // reused across all 16 N-tiles: only B-tiles stream from LDS per wmma.
  {
    const unsigned short* aRow = k_sh + (wave * 16) * SW_K;
    v16bf aT[8];
#pragma unroll
    for (int kt = 0; kt < 8; ++kt) aT[kt] = lds_tileA(aRow, SW_K, kt * 32, lane);
#pragma unroll 2
    for (int nt = 0; nt < 16; ++nt) {
      v8f acc;
#pragma unroll
      for (int r = 0; r < 8; ++r) acc[r] = 0.0f;
#pragma unroll
      for (int kt = 0; kt < 8; ++kt)
        acc = wmma_bf16(aT[kt], lds_tileB(W_t, SW_W, nt * 16, kt * 32, lane), acc);
      int mo = (lane >> 4) << 3;
      int n  = nt * 16 + (lane & 15);
      float bias = bk[n];
#pragma unroll
      for (int r = 0; r < 8; ++r)
        kz_sh[(wave * 16 + mo + r) * SW_K + n] = f2bf(acc[r] + bias);
    }
  }
  __syncthreads();

  // ---- C1: v[g] -> v_t (transposed [d][l], bf16), overwrites k_sh ----
#pragma unroll 4
  for (int i = 0; i < 128; ++i)
    v_t[t * SW_V + i] = f2bf(vg[i * 256 + t]);
  __syncthreads();

  // ---- C2: s = qz · kz^T / 16 ; wave w owns keys [16w,16w+16) ----
  {
    v8f acc;
#pragma unroll
    for (int r = 0; r < 8; ++r) acc[r] = 0.0f;
#pragma unroll
    for (int kt = 0; kt < 8; ++kt) {
      v16bf a = lds_tileA(qz_sh, SW_K, kt * 32, lane);
      v16bf b = lds_tileB(kz_sh, SW_K, wave * 16, kt * 32, lane);  // kz row = key = N
      acc = wmma_bf16(a, b, acc);
    }
    int l  = wave * 16 + (lane & 15);
    bool ok = msk[l] != 0;
    int mo = (lane >> 4) << 3;
#pragma unroll
    for (int r = 0; r < 8; ++r) {
      int M = mo + r;
      if (M < 4) s_sh[M * 128 + l] = ok ? acc[r] * 0.0625f : -__builtin_inff();
    }
  }
  __syncthreads();

  // ---- softmax over 128 keys, rows 0..3 (tiny vs matmuls) ----
  if (t < 4) {
    float mx = -__builtin_inff();
    for (int l = 0; l < 128; ++l) mx = fmaxf(mx, s_sh[t * 128 + l]);
    float sum = 0.0f;
    for (int l = 0; l < 128; ++l) {
      float e = __expf(s_sh[t * 128 + l] - mx);
      s_sh[t * 128 + l] = e;
      sum += e;
    }
    float inv = 1.0f / sum;
    for (int l = 0; l < 128; ++l)
      p_sh[t * SW_V + l] = f2bf(s_sh[t * 128 + l] * inv);
  }
  __syncthreads();

  // ---- C3: out = p @ v  (M=16 pad, N=256, K=128), 2 N-tiles/wave ----
  {
    v16bf aT[4];                      // p fragment reused across both N-tiles
#pragma unroll
    for (int kt = 0; kt < 4; ++kt) aT[kt] = lds_tileA(p_sh, SW_V, kt * 32, lane);
#pragma unroll
    for (int nn = 0; nn < 2; ++nn) {
      int nt = wave * 2 + nn;
      v8f acc;
#pragma unroll
      for (int r = 0; r < 8; ++r) acc[r] = 0.0f;
#pragma unroll
      for (int kt = 0; kt < 4; ++kt)
        acc = wmma_bf16(aT[kt], lds_tileB(v_t, SW_V, nt * 16, kt * 32, lane), acc);
      int n  = nt * 16 + (lane & 15);
      int mo = (lane >> 4) << 3;
#pragma unroll
      for (int r = 0; r < 8; ++r) {
        int M = mo + r;
        if (M < 4) out[((size_t)g * 4 + M) * 256 + n] = acc[r];
      }
    }
  }
}

extern "C" void kernel_launch(void* const* d_in, const int* in_sizes, int n_in,
                              void* d_out, int out_size, void* d_ws, size_t ws_size,
                              hipStream_t stream) {
  // setup_inputs order: 0=nq(i32), 1=q, 2=k, 3=v, 4=m(bool/u8), 5=Wq, 6=bq, 7=Wk, 8=bk
  const float*         q  = (const float*)d_in[1];
  const float*         k  = (const float*)d_in[2];
  const float*         v  = (const float*)d_in[3];
  const unsigned char* m  = (const unsigned char*)d_in[4];
  const float*         Wq = (const float*)d_in[5];
  const float*         bq = (const float*)d_in[6];
  const float*         Wk = (const float*)d_in[7];
  const float*         bk = (const float*)d_in[8];
  float* out = (float*)d_out;

  size_t lds_bytes = (size_t)LDS_USHORTS * sizeof(unsigned short);  // 295808 B
  hipLaunchKernelGGL(attn_grouped_kernel, dim3(1024), dim3(256), lds_bytes, stream,
                     q, k, v, m, Wq, bq, Wk, bk, out);
}
